// AutoregU_23794118819959
// MI455X (gfx1250) — matrix operand order repaired and checked by reading, not compiled, not run on hardware
//
#include <hip/hip_runtime.h>
#include <hip/hip_bf16.h>

// GConvGRU autoregressive rollout for MI455X (gfx1250).
// Dense paths: f16 WMMA (v_wmma_f32_16x16x32_f16), f32 accumulate.
//   - Weights are pre-packed into the exact per-lane WMMA B-fragment layout so the
//     GEMM inner loop is pure b128 loads + v_wmma (no per-element gather).
//   - One wave computes a 16x128 output strip (8 WMMAs per 32-wide k-chunk).
// Sparse Laplacian: f32 gather + global atomic adds (bandwidth bound, ~19 GB total).

typedef __attribute__((ext_vector_type(16))) _Float16 v16h;
typedef __attribute__((ext_vector_type(8)))  _Float16 v8h;
typedef __attribute__((ext_vector_type(8)))  float    v8f;

#define THREADS 256
#define HID 128
#define INF 11
#define TSTEPS 24

// ---------------- utility ----------------
__global__ void k_zero_f32(float* __restrict__ p, int n) {
  int i = blockIdx.x * blockDim.x + threadIdx.x;
  if (i < n) p[i] = 0.0f;
}

// ---------------- graph weights ----------------
__global__ void k_deg(const int* __restrict__ src, float* __restrict__ deg, int E) {
  int e = blockIdx.x * blockDim.x + threadIdx.x;
  if (e < E) atomicAdd(&deg[src[e]], 1.0f);
}

__global__ void k_edge_w(const int* __restrict__ src, const int* __restrict__ dst,
                         const float* __restrict__ deg, float* __restrict__ w, int E) {
  int e = blockIdx.x * blockDim.x + threadIdx.x;
  if (e >= E) return;
  float ds = deg[src[e]], dd = deg[dst[e]];
  float a = ds > 0.f ? rsqrtf(ds) : 0.f;
  float b = dd > 0.f ? rsqrtf(dd) : 0.f;
  w[e] = -(a * b);
}

// ---------------- weight packing (f32 -> f16, ChebConv K=2 fused) ----------------
// WxCat[g] : 32 x 128  rows 0..10 = Wx[g][0], 11..21 = Wx[g][1], 22..31 = 0
__global__ void k_pack_wx(const float* __restrict__ Wx, _Float16* __restrict__ o) {
  int gid = blockIdx.x * blockDim.x + threadIdx.x;     // 3*32*128
  if (gid >= 3 * 32 * HID) return;
  int g = gid / (32 * HID), rem = gid % (32 * HID);
  int row = rem / HID, col = rem % HID;
  float v = 0.f;
  if (row < INF)            v = Wx[((g * 2 + 0) * INF + row) * HID + col];
  else if (row < 2 * INF)   v = Wx[((g * 2 + 1) * INF + (row - INF)) * HID + col];
  o[gid] = (_Float16)v;
}

// WhCat[g] : 256 x 128  rows 0..127 = Wh[g][0], 128..255 = Wh[g][1]
__global__ void k_pack_wh(const float* __restrict__ Wh, _Float16* __restrict__ o) {
  int gid = blockIdx.x * blockDim.x + threadIdx.x;     // 3*256*128
  if (gid >= 3 * 256 * HID) return;
  int g = gid / (256 * HID), rem = gid % (256 * HID);
  int row = rem / HID, col = rem % HID;
  int k = row >> 7, i = row & 127;
  o[gid] = (_Float16)Wh[(((g * 2 + k) * HID) + i) * HID + col];
}

// Whead16 : 128 x 16 (cols 3..15 zero)
__global__ void k_pack_whead(const float* __restrict__ Wh, _Float16* __restrict__ o) {
  int gid = blockIdx.x * blockDim.x + threadIdx.x;     // 128*16
  if (gid >= HID * 16) return;
  int row = gid >> 4, c = gid & 15;
  o[gid] = (_Float16)(c < 3 ? Wh[row * 3 + c] : 0.f);
}

// Fragmentize: row-major f16 B[K x ncols] -> WMMA B-fragment layout
//   F[kc][nt][lane][j] = B[kc*32 + 16*(lane>>4) + j][nt*16 + (lane&15)]
// so each lane's v16h operand is one contiguous 32-byte block.
__global__ void k_frag_b(const _Float16* __restrict__ B, _Float16* __restrict__ F,
                         int K, int ncols) {
  int gid = blockIdx.x * blockDim.x + threadIdx.x;     // K * ncols
  if (gid >= K * ncols) return;
  int j    = gid & 15;
  int lane = (gid >> 4) & 31;
  int rest = gid >> 9;
  int nT = ncols >> 4;
  int kc = rest / nT, nt = rest - kc * nT;
  int k   = kc * 32 + ((lane >> 4) << 4) + j;
  int col = nt * 16 + (lane & 15);
  F[gid] = B[(size_t)k * ncols + col];
}

// ---------------- input assembly ----------------
__global__ void k_copy_x0(const float* __restrict__ X0, float* __restrict__ xbuf, int N) {
  int gid = blockIdx.x * blockDim.x + threadIdx.x;     // N*16
  int n = gid >> 4, c = gid & 15;
  if (n >= N) return;
  xbuf[(size_t)n * 16 + c] = (c < INF) ? X0[(size_t)n * INF + c] : 0.f;
}

// x = [X[t][:,0:3], pos, X[t][:,6:8], (pos-prev_pos)/dt]
__global__ void k_build_x(const float* __restrict__ Xt, const float* __restrict__ Xtm1,
                          const float* __restrict__ u_prev,
                          const float* __restrict__ prevp, int pstride, int poff,
                          float* __restrict__ xbuf, int N) {
  int n = blockIdx.x * blockDim.x + threadIdx.x;
  if (n >= N) return;
  const float* xr = Xt + (size_t)n * INF;
  float invdt = 1.0f / (xr[6] - Xtm1[(size_t)n * INF + 6]);
  float p0 = u_prev[(size_t)n * 3 + 0];
  float p1 = u_prev[(size_t)n * 3 + 1];
  float p2 = u_prev[(size_t)n * 3 + 2];
  const float* pp = prevp + (size_t)n * pstride + poff;
  float* o = xbuf + (size_t)n * 16;
  o[0] = xr[0]; o[1] = xr[1]; o[2] = xr[2];
  o[3] = p0;    o[4] = p1;    o[5] = p2;
  o[6] = xr[6]; o[7] = xr[7];
  o[8]  = (p0 - pp[0]) * invdt;
  o[9]  = (p1 - pp[1]) * invdt;
  o[10] = (p2 - pp[2]) * invdt;
}

// ---------------- sparse Laplacian ----------------
__global__ void k_scatter_lapx(const int* __restrict__ src, const int* __restrict__ dst,
                               const float* __restrict__ w, const float* __restrict__ x,
                               float* __restrict__ lapx, int E) {
  int gid = blockIdx.x * blockDim.x + threadIdx.x;
  int e = gid >> 4, f = gid & 15;
  if (e >= E || f >= INF) return;
  atomicAdd(lapx + (size_t)dst[e] * 16 + f, w[e] * x[(size_t)src[e] * 16 + f]);
}

__global__ void k_scatter_lap128(const int* __restrict__ src, const int* __restrict__ dst,
                                 const float* __restrict__ w, const float* __restrict__ x,
                                 float* __restrict__ lap, int E) {
  long long gid = (long long)blockIdx.x * blockDim.x + threadIdx.x;
  long long e = gid >> 5;
  int q = (int)(gid & 31);
  if (e >= E) return;
  int s = src[e], d = dst[e];
  float we = w[e];
  const float4 xv = *(const float4*)(x + (size_t)s * HID + q * 4);
  float* lp = lap + (size_t)d * HID + q * 4;
  atomicAdd(lp + 0, we * xv.x);
  atomicAdd(lp + 1, we * xv.y);
  atomicAdd(lp + 2, we * xv.z);
  atomicAdd(lp + 3, we * xv.w);
}

// ---------------- f16 staging ----------------
__global__ void k_pack_xcat(const float* __restrict__ xbuf, const float* __restrict__ lapx,
                            _Float16* __restrict__ xcat, int N) {
  int gid = blockIdx.x * blockDim.x + threadIdx.x;     // N*32
  int n = gid >> 5, c = gid & 31;
  if (n >= N) return;
  float v = 0.f;
  if (c < INF)          v = xbuf[(size_t)n * 16 + c];
  else if (c < 2 * INF) v = lapx[(size_t)n * 16 + (c - INF)];
  xcat[(size_t)n * 32 + c] = (_Float16)v;
}

__global__ void k_pack_lap_hcat(const float* __restrict__ lap, _Float16* __restrict__ hcat,
                                int total) {
  int i = blockIdx.x * blockDim.x + threadIdx.x;
  if (i >= total) return;
  int n = i >> 7, c = i & 127;
  hcat[(size_t)n * 256 + HID + c] = (_Float16)lap[i];
}

// ---------------- WMMA GEMM ----------------
// One wave -> 16 rows x (NT*16) cols.  A[M x K](f16, ld=lda) row-major;
// BF = pre-fragmentized weights ([K/32][NT][32][16] halfs).  K multiple of 32.
template <int NT>
__global__ __launch_bounds__(THREADS)
void k_wmma_gemm(const _Float16* __restrict__ A, int lda,
                 const _Float16* __restrict__ BF,
                 const float* __restrict__ bias,
                 float* __restrict__ C, int ldc, int vcols,
                 int M, int K) {
  int wave = (blockIdx.x * blockDim.x + threadIdx.x) >> 5;
  int lane = threadIdx.x & 31;
  if (wave * 16 >= M) return;               // wave-uniform: EXEC all-ones for WMMA
  int hi = lane >> 4, lm = lane & 15;

  const _Float16* arow = A + (size_t)(wave * 16 + lm) * lda + hi * 16;
  const _Float16* bf   = BF + (size_t)lane * 16;

  v8f acc[NT];
#pragma unroll
  for (int nt = 0; nt < NT; ++nt) acc[nt] = {};

  const int nchunks = K >> 5;
  for (int kc = 0; kc < nchunks; ++kc) {
    union { v16h v; v8h h[2]; } au;
    au.h[0] = *(const v8h*)(arow + kc * 32);        // A[row][k0 + 16*hi + 0..7]
    au.h[1] = *(const v8h*)(arow + kc * 32 + 8);    // A[row][k0 + 16*hi + 8..15]
#pragma unroll
    for (int nt = 0; nt < NT; ++nt) {
      v16h b = *(const v16h*)(bf + ((size_t)(kc * NT + nt) << 9));  // lane-contig 32B
      acc[nt] = __builtin_amdgcn_wmma_f32_16x16x32_f16(false, au.v, false, b,
                                                       (short)0, acc[nt], false, false);
    }
  }

#pragma unroll
  for (int nt = 0; nt < NT; ++nt) {
    int col = nt * 16 + lm;
    if (col < vcols) {
      float bv = bias ? bias[col] : 0.f;
#pragma unroll
      for (int v = 0; v < 8; ++v) {         // D: row = v + 8*hi, col = lane&15
        int row = wave * 16 + v + hi * 8;
        if (row < M) C[(size_t)row * ldc + col] = acc[nt][v] + bv;
      }
    }
  }
}

// ---------------- GRU elementwise ----------------
__global__ void k_gates_zr(float* __restrict__ gx0, const float* __restrict__ gh0,
                           float* __restrict__ gx1, const float* __restrict__ gh1,
                           const float* __restrict__ h, _Float16* __restrict__ hcat,
                           int total) {
  int i = blockIdx.x * blockDim.x + threadIdx.x;
  if (i >= total) return;
  float z = 1.f / (1.f + expf(-(gx0[i] + gh0[i])));
  float r = 1.f / (1.f + expf(-(gx1[i] + gh1[i])));
  float hr = h[i] * r;
  gx0[i] = z;
  gx1[i] = hr;
  int n = i >> 7, c = i & 127;
  hcat[(size_t)n * 256 + c] = (_Float16)hr;
}

__global__ void k_update_h(const float* __restrict__ gx2, const float* __restrict__ ghc,
                           const float* __restrict__ z, float* __restrict__ h,
                           _Float16* __restrict__ hcat, int total) {
  int i = blockIdx.x * blockDim.x + threadIdx.x;
  if (i >= total) return;
  float ht = tanhf(gx2[i] + ghc[i]);
  float zv = z[i];
  float hn = zv * h[i] + (1.f - zv) * ht;
  h[i] = hn;
  int n = i >> 7, c = i & 127;
  hcat[(size_t)n * 256 + c] = (_Float16)hn;
}

// ---------------- host launcher ----------------
static inline int gs(long long n) { return (int)((n + THREADS - 1) / THREADS); }

extern "C" void kernel_launch(void* const* d_in, const int* in_sizes, int n_in,
                              void* d_out, int out_size, void* d_ws, size_t ws_size,
                              hipStream_t stream) {
  const float* X     = (const float*)d_in[0];   // [T,N,11]
  const int*   ei    = (const int*)d_in[1];     // [2,E]
  const float* Wx    = (const float*)d_in[2];
  const float* bx    = (const float*)d_in[3];
  const float* Wh    = (const float*)d_in[4];
  const float* bh    = (const float*)d_in[5];
  const float* Whead = (const float*)d_in[6];
  const float* bhead = (const float*)d_in[7];
  float* out = (float*)d_out;

  const int N = in_sizes[0] / (TSTEPS * INF);
  const int E = in_sizes[1] / 2;
  const int* src = ei;
  const int* dst = ei + E;
  const int NH = N * HID;

  // workspace layout (float units, 256B-aligned slices)
  float* ws = (float*)d_ws;
  size_t off = 0;
  auto alloc = [&](size_t n) { size_t o = off; off += (n + 63) & ~(size_t)63; return o; };
  float* deg   = ws + alloc(N);
  float* wE    = ws + alloc(E);
  float* h     = ws + alloc(NH);
  float* lap   = ws + alloc(NH);
  float* gx0   = ws + alloc(NH);   // later holds z
  float* gx1   = ws + alloc(NH);   // later holds h*r
  float* gx2   = ws + alloc(NH);
  float* gh0   = ws + alloc(NH);   // later holds candidate preact
  float* gh1   = ws + alloc(NH);
  float* xbuf  = ws + alloc((size_t)N * 16);
  float* lapx  = ws + alloc((size_t)N * 16);
  _Float16* hcat  = (_Float16*)(ws + alloc(NH));             // N x 256 halfs
  _Float16* xcat  = (_Float16*)(ws + alloc((size_t)N * 16)); // N x 32 halfs
  _Float16* wxcat = (_Float16*)(ws + alloc(3 * 32 * HID / 2));   // row-major temps
  _Float16* whcat = (_Float16*)(ws + alloc(3 * 256 * HID / 2));
  _Float16* whd16 = (_Float16*)(ws + alloc(HID * 16 / 2));
  _Float16* wxfrag = (_Float16*)(ws + alloc(3 * 32 * HID / 2));  // fragmentized
  _Float16* whfrag = (_Float16*)(ws + alloc(3 * 256 * HID / 2));
  _Float16* whdfrag = (_Float16*)(ws + alloc(HID * 16 / 2));

  // ---- one-time precompute (deterministic, repeated every call) ----
  k_zero_f32<<<gs(N), THREADS, 0, stream>>>(deg, N);
  k_deg<<<gs(E), THREADS, 0, stream>>>(src, deg, E);
  k_edge_w<<<gs(E), THREADS, 0, stream>>>(src, dst, deg, wE, E);
  k_pack_wx<<<gs(3 * 32 * HID), THREADS, 0, stream>>>(Wx, wxcat);
  k_pack_wh<<<gs(3 * 256 * HID), THREADS, 0, stream>>>(Wh, whcat);
  k_pack_whead<<<gs(HID * 16), THREADS, 0, stream>>>(Whead, whd16);
  for (int g = 0; g < 3; ++g) {
    k_frag_b<<<gs(32 * HID), THREADS, 0, stream>>>(
        wxcat + (size_t)g * 32 * HID, wxfrag + (size_t)g * 32 * HID, 32, HID);
    k_frag_b<<<gs(256 * HID), THREADS, 0, stream>>>(
        whcat + (size_t)g * 256 * HID, whfrag + (size_t)g * 256 * HID, 256, HID);
  }
  k_frag_b<<<gs(HID * 16), THREADS, 0, stream>>>(whd16, whdfrag, HID, 16);
  k_zero_f32<<<gs(NH), THREADS, 0, stream>>>(h, NH);
  k_zero_f32<<<gs(NH), THREADS, 0, stream>>>((float*)hcat, NH);  // N*256 halfs

  const int mtiles = (N + 15) / 16;
  const int gemmBlocks = gs((long long)mtiles * 32);  // one wave per 16-row strip

  for (int t = 0; t < TSTEPS; ++t) {
    // -- assemble x_t --
    if (t == 0) {
      k_copy_x0<<<gs((long long)N * 16), THREADS, 0, stream>>>(X, xbuf, N);
    } else {
      const float* Xt    = X + (size_t)t * N * INF;
      const float* Xtm1  = X + (size_t)(t - 1) * N * INF;
      const float* uprev = out + (size_t)(t - 1) * N * 3;
      const float* prevp;
      int pstride, poff;
      if (t == 1) { prevp = X; pstride = INF; poff = 3; }            // X[0][:,3:6]
      else        { prevp = out + (size_t)(t - 2) * N * 3; pstride = 3; poff = 0; }
      k_build_x<<<gs(N), THREADS, 0, stream>>>(Xt, Xtm1, uprev, prevp, pstride, poff, xbuf, N);
    }
    // lap(x)
    k_zero_f32<<<gs((long long)N * 16), THREADS, 0, stream>>>(lapx, N * 16);
    k_scatter_lapx<<<gs((long long)E * 16), THREADS, 0, stream>>>(src, dst, wE, xbuf, lapx, E);
    k_pack_xcat<<<gs((long long)N * 32), THREADS, 0, stream>>>(xbuf, lapx, xcat, N);
    // gx[g] = [x|lap x] @ WxCat[g] + bx[g]   (K=32 WMMA)
    float* gxp[3] = {gx0, gx1, gx2};
    for (int g = 0; g < 3; ++g)
      k_wmma_gemm<8><<<gemmBlocks, THREADS, 0, stream>>>(
          xcat, 32, wxfrag + (size_t)g * 32 * HID, bx + g * HID,
          gxp[g], HID, HID, N, 32);

    // lap(h) -> hcat[:, 128:256]
    k_zero_f32<<<gs(NH), THREADS, 0, stream>>>(lap, NH);
    k_scatter_lap128<<<gs((long long)E * 32), THREADS, 0, stream>>>(src, dst, wE, h, lap, E);
    k_pack_lap_hcat<<<gs(NH), THREADS, 0, stream>>>(lap, hcat, NH);
    // gh[g] = [h|lap h] @ WhCat[g] + bh[g]   (K=256 WMMA), g = 0 (z), 1 (r)
    k_wmma_gemm<8><<<gemmBlocks, THREADS, 0, stream>>>(
        hcat, 256, whfrag + (size_t)0 * 256 * HID, bh + 0 * HID, gh0, HID, HID, N, 256);
    k_wmma_gemm<8><<<gemmBlocks, THREADS, 0, stream>>>(
        hcat, 256, whfrag + (size_t)1 * 256 * HID, bh + 1 * HID, gh1, HID, HID, N, 256);

    // z, r, h*r
    k_gates_zr<<<gs(NH), THREADS, 0, stream>>>(gx0, gh0, gx1, gh1, h, hcat, NH);

    // lap(h*r) -> hcat[:, 128:256]; candidate preact into gh0
    k_zero_f32<<<gs(NH), THREADS, 0, stream>>>(lap, NH);
    k_scatter_lap128<<<gs((long long)E * 32), THREADS, 0, stream>>>(src, dst, wE, gx1, lap, E);
    k_pack_lap_hcat<<<gs(NH), THREADS, 0, stream>>>(lap, hcat, NH);
    k_wmma_gemm<8><<<gemmBlocks, THREADS, 0, stream>>>(
        hcat, 256, whfrag + (size_t)2 * 256 * HID, bh + 2 * HID, gh0, HID, HID, N, 256);

    // h update (writes f32 h + f16 into hcat[:, :128])
    k_update_h<<<gs(NH), THREADS, 0, stream>>>(gx2, gh0, gx0, h, hcat, NH);

    // head: u_t = h @ W_head + b_head  (K=128 WMMA over hcat[:, :128], 3 valid cols)
    k_wmma_gemm<1><<<gemmBlocks, THREADS, 0, stream>>>(
        hcat, 256, whdfrag, bhead, out + (size_t)t * N * 3, 3, 3, N, 128);
  }
}
